// Self_Attn_8907762172299
// MI455X (gfx1250) — matrix operand order repaired and checked by reading, not compiled, not run on hardware
//
#include <hip/hip_runtime.h>

// Problem constants (match reference)
#define B_SZ   4
#define C_CH   256
#define N_SP   4096   // 64*64
#define WH     64
#define DK     32

typedef __attribute__((ext_vector_type(16))) _Float16 v16h;
typedef __attribute__((ext_vector_type(8)))  float    v8f;
typedef __attribute__((ext_vector_type(4)))  float    v4f;

// Four adjacent 16x16 f32 tiles of  D = A[arow0:+16, 0:Kc] * X[0:Kc, n0:n0+64]
// computed with f16-split WMMA (hi/lo decomposition -> ~fp32 accuracy).
// A-fragments (weights) are built once per K-step and reused across the 4
// N-tiles, so the cvt/VALU cost is amortized 4x and WMMA density is high.
// A row-major [*, lda], X row-major [Kc, ldx]. Whole wave participates.
__device__ __forceinline__ void wmma_tile_split4(const float* __restrict__ A, int lda, int arow0,
                                                 const float* __restrict__ X, int ldx, int n0,
                                                 int Kc, v8f acc[4])
{
  const int lane = threadIdx.x & 31;
  const int l  = lane & 15;
  const int hi = lane >> 4;
  for (int k0 = 0; k0 < Kc; k0 += 32) {
    v16h ah = {}, al = {};
#pragma unroll
    for (int r = 0; r < 8; ++r) {
      // A fragment: 16-bit A 16x32 layout (ISA 7.12.2)
      const int ka = (r < 4) ? (hi * 8 + 2 * r) : (16 + hi * 8 + 2 * (r - 4));
      const float* ap = A + (size_t)(arow0 + l) * lda + k0 + ka;
      const float a0 = ap[0], a1 = ap[1];
      const _Float16 h0 = (_Float16)a0, h1 = (_Float16)a1;
      ah[2 * r]     = h0;
      ah[2 * r + 1] = h1;
      al[2 * r]     = (_Float16)(a0 - (float)h0);
      al[2 * r + 1] = (_Float16)(a1 - (float)h1);
    }
#pragma unroll
    for (int t = 0; t < 4; ++t) {
      const int nt = n0 + 16 * t;
      v16h bh = {}, bl = {};
#pragma unroll
      for (int r = 0; r < 8; ++r) {
        // B fragment: 32x16, same K order as A (r<4: hi*8 + 2r, 2r+1 ; r>=4: 16 + hi*8 + 2(r-4), +1) ; N = l
        const int kb = (r < 4) ? (hi * 8 + 2 * r) : (16 + hi * 8 + 2 * (r - 4));
        const float b0 = X[(size_t)(k0 + kb)     * ldx + nt + l];
        const float b1 = X[(size_t)(k0 + kb + 1) * ldx + nt + l];
        const _Float16 g0 = (_Float16)b0, g1 = (_Float16)b1;
        bh[2 * r]     = g0;
        bh[2 * r + 1] = g1;
        bl[2 * r]     = (_Float16)(b0 - (float)g0);
        bl[2 * r + 1] = (_Float16)(b1 - (float)g1);
      }
      acc[t] = __builtin_amdgcn_wmma_f32_16x16x32_f16(false, ah, false, bh, (short)0, acc[t], false, false);
      acc[t] = __builtin_amdgcn_wmma_f32_16x16x32_f16(false, al, false, bh, (short)0, acc[t], false, false);
      acc[t] = __builtin_amdgcn_wmma_f32_16x16x32_f16(false, ah, false, bl, (short)0, acc[t], false, false);
    }
  }
}

// Stage 1: qk[b][n][0:32] = q(n), qk[b][n][32:64] = k(n)   (row-major per position)
// Combined 64x256 weight: mtile 0,1 -> Wq ; mtile 2,3 -> Wk.
__global__ void qk_kernel(const float* __restrict__ x,
                          const float* __restrict__ Wq, const float* __restrict__ bq,
                          const float* __restrict__ Wk, const float* __restrict__ bk,
                          float* __restrict__ qkbuf)
{
  const int wid   = threadIdx.x >> 5;
  const int n0    = (blockIdx.x * 8 + wid) * 64;   // 4 tiles of 16 per wave
  const int mtile = blockIdx.y;                    // 0..3 (uniform per block)
  const int b     = blockIdx.z;
  const float* A    = (mtile < 2) ? Wq : Wk;
  const float* bias = (mtile < 2) ? bq : bk;
  const int m0l = (mtile & 1) * 16;                // row offset inside Wq/Wk
  const float* X = x + (size_t)b * C_CH * N_SP;

  v8f acc[4] = {};
  wmma_tile_split4(A, C_CH, m0l, X, N_SP, n0, C_CH, acc);

  const int lane = threadIdx.x & 31;
  const int l = lane & 15, hi = lane >> 4;
#pragma unroll
  for (int t = 0; t < 4; ++t) {
#pragma unroll
    for (int r = 0; r < 8; ++r) {
      const int dloc  = m0l + r + 8 * hi;          // row within Wq/Wk  (0..31)
      const int dcomb = mtile * 16 + r + 8 * hi;   // combined d index (0..63)
      const int n = n0 + 16 * t + l;
      qkbuf[((size_t)b * N_SP + n) * 64 + dcomb] = acc[t][r] + bias[dloc];
    }
  }
}

// Stage 2: v[b][c][n] = Wv . x + bv     ([C x N] row-major per batch)
__global__ void v_kernel(const float* __restrict__ x,
                         const float* __restrict__ Wv, const float* __restrict__ bv,
                         float* __restrict__ vbuf)
{
  const int wid = threadIdx.x >> 5;
  const int n0  = (blockIdx.x * 8 + wid) * 64;     // 4 tiles of 16 per wave
  const int m0  = blockIdx.y * 16;
  const int b   = blockIdx.z;
  const float* X = x + (size_t)b * C_CH * N_SP;

  v8f acc[4] = {};
  wmma_tile_split4(Wv, C_CH, m0, X, N_SP, n0, C_CH, acc);

  const int lane = threadIdx.x & 31;
  const int l = lane & 15, hi = lane >> 4;
#pragma unroll
  for (int t = 0; t < 4; ++t) {
#pragma unroll
    for (int r = 0; r < 8; ++r) {
      const int m = m0 + r + 8 * hi;
      vbuf[(size_t)b * C_CH * N_SP + (size_t)m * N_SP + n0 + 16 * t + l] = acc[t][r] + bv[m];
    }
  }
}

// Stage 3: per (b, i): 9 window scores -> masked softmax -> w[9];
// write w to wbuf and stream the full dense attention row (zeros + scatter)
// with NON-TEMPORAL stores: 268 MB write-once output must not thrash the
// 192 MB L2 where x / qk / v stay resident for stage 4.
__global__ void attn_kernel(const float* __restrict__ qkbuf,
                            float* __restrict__ wbuf,
                            float* __restrict__ attn)
{
  const int i = blockIdx.x;       // position 0..4095
  const int b = blockIdx.y;
  const int t = threadIdx.x;      // 0..255

  __shared__ float s[9];
  __shared__ float wv[9];
  __shared__ int   jidx[9];
  __shared__ int   jval[9];

  const int r = i >> 6, c = i & 63;

  if (t < 9) {
    const int dr = t / 3 - 1, dc = t % 3 - 1;
    const int rr = r + dr, cc = c + dc;
    const int valid = (rr >= 0 && rr < WH && cc >= 0 && cc < WH);
    const int rrc = min(max(rr, 0), WH - 1);
    const int ccc = min(max(cc, 0), WH - 1);
    const int j = rrc * WH + ccc;
    jidx[t] = j;
    jval[t] = valid;
    const float* qrow = qkbuf + ((size_t)b * N_SP + i) * 64;
    const float* krow = qkbuf + ((size_t)b * N_SP + j) * 64 + 32;
    float sc = 0.0f;
#pragma unroll
    for (int d = 0; d < DK; ++d) sc += qrow[d] * krow[d];
    s[t] = valid ? sc : -3.0e30f;
  }
  __syncthreads();

  if (t == 0) {
    float m = s[0];
#pragma unroll
    for (int k = 1; k < 9; ++k) m = fmaxf(m, s[k]);
    float e[9];
    float sum = 0.0f;
#pragma unroll
    for (int k = 0; k < 9; ++k) { e[k] = jval[k] ? expf(s[k] - m) : 0.0f; sum += e[k]; }
    const float inv = 1.0f / sum;
#pragma unroll
    for (int k = 0; k < 9; ++k) wv[k] = e[k] * inv;
  }
  __syncthreads();

  if (t < 9) wbuf[((size_t)b * N_SP + i) * 9 + t] = wv[t];

  // Write 16 columns of this attention row per thread: zeros + scattered w.
  float vals[16];
#pragma unroll
  for (int u = 0; u < 16; ++u) vals[u] = 0.0f;
  const int col0 = t * 16;
#pragma unroll
  for (int k = 0; k < 9; ++k) {
    const int j = jidx[k];
    if (jval[k] && j >= col0 && j < col0 + 16) vals[j - col0] += wv[k];
  }
  v4f* dst = (v4f*)(attn + ((size_t)b * N_SP + i) * N_SP + col0);
#pragma unroll
  for (int u = 0; u < 4; ++u) {
    v4f pkt = {vals[4 * u], vals[4 * u + 1], vals[4 * u + 2], vals[4 * u + 3]};
    __builtin_nontemporal_store(pkt, dst + u);   // TH=NT streaming store
  }
}

// Stage 4: out[b][c][i] = gamma * sum_k w[b][i][k] * v[b][c][nbr(i,k)] + x[b][c][i]
__global__ void out_kernel(const float* __restrict__ x,
                           const float* __restrict__ vbuf,
                           const float* __restrict__ wbuf,
                           const float* __restrict__ gamma,
                           float* __restrict__ out)
{
  const int i  = blockIdx.x * blockDim.x + threadIdx.x;  // 0..4095
  const int ch = blockIdx.y;
  const int b  = blockIdx.z;
  const int r = i >> 6, c = i & 63;

  const float* wrow = wbuf + ((size_t)b * N_SP + i) * 9;
  const float* vrow = vbuf + (size_t)b * C_CH * N_SP + (size_t)ch * N_SP;

  float acc = 0.0f;
#pragma unroll
  for (int k = 0; k < 9; ++k) {
    const int dr = k / 3 - 1, dc = k % 3 - 1;
    const int rr = min(max(r + dr, 0), WH - 1);
    const int cc = min(max(c + dc, 0), WH - 1);
    acc += wrow[k] * vrow[rr * WH + cc];   // invalid taps have w == 0
  }
  const size_t o = (size_t)b * C_CH * N_SP + (size_t)ch * N_SP + i;
  out[o] = gamma[0] * acc + x[o];
}

extern "C" void kernel_launch(void* const* d_in, const int* in_sizes, int n_in,
                              void* d_out, int out_size, void* d_ws, size_t ws_size,
                              hipStream_t stream)
{
  (void)in_sizes; (void)n_in; (void)out_size; (void)ws_size;
  const float* x     = (const float*)d_in[0];
  const float* Wq    = (const float*)d_in[1];
  const float* bq    = (const float*)d_in[2];
  const float* Wk    = (const float*)d_in[3];
  const float* bk    = (const float*)d_in[4];
  const float* Wv    = (const float*)d_in[5];
  const float* bv    = (const float*)d_in[6];
  const float* gamma = (const float*)d_in[7];

  float* out  = (float*)d_out;
  float* attn = out + (size_t)B_SZ * C_CH * N_SP;   // outputs concatenated: out, attention

  // Workspace layout (floats): qk [B,N,64] | v [B,C,N] | w [B,N,9]  (~20.6 MB)
  float* qkbuf = (float*)d_ws;
  float* vbuf  = qkbuf + (size_t)B_SZ * N_SP * 64;
  float* wbuf  = vbuf  + (size_t)B_SZ * C_CH * N_SP;

  // Stage 1: q||k channel GEMM (M=64, K=256, N=4096 per batch), WMMA.
  // Block = 8 waves, each wave does 4 N-tiles -> block covers 512 columns.
  qk_kernel<<<dim3(N_SP / 512, 4, B_SZ), 256, 0, stream>>>(x, Wq, bq, Wk, bk, qkbuf);
  // Stage 2: V channel GEMM (M=256, K=256, N=4096 per batch), WMMA.
  v_kernel<<<dim3(N_SP / 512, C_CH / 16, B_SZ), 256, 0, stream>>>(x, Wv, bv, vbuf);
  // Stage 3: scores + softmax + stream full dense attention rows (NT stores).
  attn_kernel<<<dim3(N_SP, B_SZ, 1), 256, 0, stream>>>(qkbuf, wbuf, attn);
  // Stage 4: gather-weighted V + residual.
  out_kernel<<<dim3(N_SP / 256, C_CH, B_SZ), 256, 0, stream>>>(x, vbuf, wbuf, gamma, out);
}
